// ImprovedDGI_6648609374955
// MI455X (gfx1250) — matrix operand order repaired and checked by reading, not compiled, not run on hardware
//
#include <hip/hip_runtime.h>

#define DD   256
#define HH2  512
#define TEMP 0.5f
#define EPSV 1e-5f
#define NPART 512

typedef __attribute__((ext_vector_type(16))) __bf16 v16bf;
typedef __attribute__((ext_vector_type(8)))  __bf16 v8bf;
typedef __attribute__((ext_vector_type(8)))  float  v8f;

static __device__ __forceinline__ unsigned short f2bf(float f) {
    union { float f; unsigned u; } v; v.f = f;
    unsigned r = v.u + 0x7FFFu + ((v.u >> 16) & 1u);   // round-to-nearest-even
    return (unsigned short)(r >> 16);
}

// ---------------- utility kernels ----------------

__global__ void k_zero_f32(float* p, long long n) {
    long long i = (long long)blockIdx.x * blockDim.x + threadIdx.x;
    long long stride = (long long)gridDim.x * blockDim.x;
    for (; i < n; i += stride) p[i] = 0.0f;
}

// convert all weights to bf16, transposed to [Dout][K] row-major
__global__ void k_conv_weights(const float* Wg, const float* Wd1, const float* Wd2,
                               const float* Wd3, unsigned short* WgT,
                               unsigned short* Wd1T, unsigned short* Wd2T,
                               unsigned short* W3b) {
    int total = DD*DD + HH2*DD + DD*HH2 + DD;
    for (int idx = blockIdx.x * blockDim.x + threadIdx.x; idx < total;
         idx += gridDim.x * blockDim.x) {
        int j = idx;
        if (j < DD*DD) {                       // WgT[n*DD+k] = Wg[k*DD+n]
            int n = j / DD, k = j % DD;
            WgT[j] = f2bf(Wg[k*DD + n]);
        } else if ((j -= DD*DD) < HH2*DD) {    // Wd1T[n*DD+k] = Wd1[k*HH2+n], n<HH2
            int n = j / DD, k = j % DD;
            Wd1T[j] = f2bf(Wd1[k*HH2 + n]);
        } else if ((j -= HH2*DD) < DD*HH2) {   // Wd2T[n*HH2+k] = Wd2[k*DD+n], n<DD
            int n = j / HH2, k = j % HH2;
            Wd2T[j] = f2bf(Wd2[k*DD + n]);
        } else {
            j -= DD*HH2;
            W3b[j] = f2bf(Wd3[j]);
        }
    }
}

// ---------------- graph aggregation ----------------

__global__ void k_edge_deg(const int* ei, float* deg, long long E) {
    long long t = (long long)blockIdx.x * blockDim.x + threadIdx.x;
    if (t < E) atomicAdd(deg + ei[E + t], 1.0f);     // dst row
}

__global__ void k_edge_scatter(const int* ei, const float* x, float* agg, long long E) {
    long long t = (long long)blockIdx.x * blockDim.x + threadIdx.x;
    if (t >= E * 64) return;
    long long e = t >> 6;
    int c = (int)(t & 63) << 2;
    int s = ei[e];
    int d = ei[E + e];
    const float4 v = *(const float4*)(x + (long long)s * DD + c);
    float* p = agg + (long long)d * DD + c;
    atomicAdd(p + 0, v.x); atomicAdd(p + 1, v.y);
    atomicAdd(p + 2, v.z); atomicAdd(p + 3, v.w);
}

// m = x + agg/max(deg,1)  -> bf16
__global__ void k_prep_m(const float* x, const float* agg, const float* deg,
                         unsigned short* mbf, long long total) {
    long long i = (long long)blockIdx.x * blockDim.x + threadIdx.x;
    if (i >= total) return;
    long long n = i >> 8;
    float dg = fmaxf(deg[n], 1.0f);
    mbf[i] = f2bf(x[i] + agg[i] / dg);
}

// ---------------- bf16 WMMA GEMM ----------------
// out[M,Dout] = epi(A[M,K] @ Wt[Dout,K]^T + bias)
// One wave computes a 32x64 output tile: 2 A frags x 4 B frags = 8 WMMAs / k-step.
// M % 32 == 0, Dout % 64 == 0, K % 32 == 0 required (holds: M=200000, Dout in {256,512}).

__global__ __launch_bounds__(256)
void k_wmma_gemm(const unsigned short* A_, const unsigned short* Wt_,
                 const float* bias, float* out, int Mrows, int K, int Dout, int relu) {
    int wave    = blockIdx.x * (blockDim.x >> 5) + (threadIdx.x >> 5);
    int tiles_n = Dout >> 6;                           // 64-wide groups
    int total   = (Mrows >> 5) * tiles_n;
    if (wave >= total) return;                         // uniform per wave: EXEC stays all-ones
    int tm = wave / tiles_n, tn = wave % tiles_n;
    int lane = threadIdx.x & 31;
    int half = lane >> 4, l = lane & 15;

    const __bf16* A  = (const __bf16*)A_;
    const __bf16* Wt = (const __bf16*)Wt_;
    // A frag rows: lane holds K {0..15} (half=0) / {16..31} (half=1)
    long long arow0 = (long long)(tm * 32 + l) * K + half * 16;
    long long arow1 = arow0 + (long long)16 * K;
    // B frag cols: lane holds K {0..15} (half=0) / {16..31} (half=1)
    long long brow[4];
#pragma unroll
    for (int g = 0; g < 4; g++)
        brow[g] = (long long)(tn * 64 + g * 16 + l) * K + half * 16;

    v8f acc[2][4] = {};
    for (int k0 = 0; k0 < K; k0 += 32) {
        v8bf a00 = *(const v8bf*)(A + arow0 + k0);
        v8bf a01 = *(const v8bf*)(A + arow0 + k0 + 8);
        v8bf a10 = *(const v8bf*)(A + arow1 + k0);
        v8bf a11 = *(const v8bf*)(A + arow1 + k0 + 8);
        v16bf a0 = __builtin_shufflevector(a00, a01, 0,1,2,3,4,5,6,7,8,9,10,11,12,13,14,15);
        v16bf a1 = __builtin_shufflevector(a10, a11, 0,1,2,3,4,5,6,7,8,9,10,11,12,13,14,15);
#pragma unroll
        for (int g = 0; g < 4; g++) {
            v16bf b = *(const v16bf*)(Wt + brow[g] + k0);
            acc[0][g] = __builtin_amdgcn_wmma_f32_16x16x32_bf16(false, a0, false, b,
                                                                (short)0, acc[0][g], false, false);
            acc[1][g] = __builtin_amdgcn_wmma_f32_16x16x32_bf16(false, a1, false, b,
                                                                (short)0, acc[1][g], false, false);
        }
    }
#pragma unroll
    for (int g = 0; g < 4; g++) {
        int col = tn * 64 + g * 16 + l;
        float bv = bias[col];
#pragma unroll
        for (int r = 0; r < 2; r++) {
#pragma unroll
            for (int i = 0; i < 8; i++) {
                int row = tm * 32 + r * 16 + i + half * 8;   // C/D layout: VGPR i -> rows i / i+8
                float v = acc[r][g][i] + bv;
                if (relu) v = fmaxf(v, 0.0f);
                out[(long long)row * Dout + col] = v;
            }
        }
    }
}

// ---------------- attention pooling ----------------

__global__ void k_row_logits(const float* h, float* logits, int N) {
    int n = blockIdx.x * blockDim.x + threadIdx.x;
    if (n >= N) return;
    const float4* p = (const float4*)(h + (long long)n * DD);
    float s = 0.0f;
    for (int i = 0; i < DD / 4; i++) { float4 v = p[i]; s += v.x + v.y + v.z + v.w; }
    logits[n] = s * (1.0f / DD) * (1.0f / TEMP);
}

__global__ void k_partial_max(const float* logits, float* part, int N) {
    __shared__ float sh[256];
    float m = -3.4e38f;
    for (long long i = (long long)blockIdx.x * blockDim.x + threadIdx.x; i < N;
         i += (long long)gridDim.x * blockDim.x)
        m = fmaxf(m, logits[i]);
    sh[threadIdx.x] = m; __syncthreads();
    for (int s = 128; s > 0; s >>= 1) {
        if ((int)threadIdx.x < s) sh[threadIdx.x] = fmaxf(sh[threadIdx.x], sh[threadIdx.x + s]);
        __syncthreads();
    }
    if (threadIdx.x == 0) part[blockIdx.x] = sh[0];
}

__global__ void k_final_max(const float* part, float* scal) {
    __shared__ float sh[256];
    sh[threadIdx.x] = fmaxf(part[threadIdx.x], part[threadIdx.x + 256]);
    __syncthreads();
    for (int s = 128; s > 0; s >>= 1) {
        if ((int)threadIdx.x < s) sh[threadIdx.x] = fmaxf(sh[threadIdx.x], sh[threadIdx.x + s]);
        __syncthreads();
    }
    if (threadIdx.x == 0) scal[0] = sh[0];
}

__global__ void k_partial_sumexp(const float* logits, const float* scal, float* part, int N) {
    __shared__ float sh[256];
    float mx = scal[0], s = 0.0f;
    for (long long i = (long long)blockIdx.x * blockDim.x + threadIdx.x; i < N;
         i += (long long)gridDim.x * blockDim.x)
        s += __expf(logits[i] - mx);
    sh[threadIdx.x] = s; __syncthreads();
    for (int t = 128; t > 0; t >>= 1) {
        if ((int)threadIdx.x < t) sh[threadIdx.x] += sh[threadIdx.x + t];
        __syncthreads();
    }
    if (threadIdx.x == 0) part[blockIdx.x] = sh[0];
}

__global__ void k_final_sumexp(const float* part, float* scal) {
    __shared__ float sh[256];
    sh[threadIdx.x] = part[threadIdx.x] + part[threadIdx.x + 256];
    __syncthreads();
    for (int s = 128; s > 0; s >>= 1) {
        if ((int)threadIdx.x < s) sh[threadIdx.x] += sh[threadIdx.x + s];
        __syncthreads();
    }
    if (threadIdx.x == 0) scal[1] = 1.0f / sh[0];
}

// summary0[d] = sum_n softmax(logits)[n] * h[n,d]
__global__ void k_col_wsum(const float* h, const float* logits, const float* scal,
                           float* summary0, int N) {
    __shared__ float sh[256];
    int d = blockIdx.x;
    float mx = scal[0], inv = scal[1], s = 0.0f;
    for (int n = threadIdx.x; n < N; n += blockDim.x)
        s += __expf(logits[n] - mx) * h[(long long)n * DD + d];
    sh[threadIdx.x] = s; __syncthreads();
    for (int t = 128; t > 0; t >>= 1) {
        if ((int)threadIdx.x < t) sh[threadIdx.x] += sh[threadIdx.x + t];
        __syncthreads();
    }
    if (threadIdx.x == 0) summary0[d] = sh[0] * inv;
}

// summary MLP: relu(s@Ws1+bs1)@Ws2+bs2  (single block, 256 threads)
__global__ void k_summary_mlp(const float* Ws1, const float* bs1, const float* Ws2,
                              const float* bs2, const float* summary0, float* summary2) {
    __shared__ float s0[DD], s1[DD];
    int t = threadIdx.x;
    s0[t] = summary0[t];
    __syncthreads();
    float a = bs1[t];
    for (int k = 0; k < DD; k++) a += s0[k] * Ws1[k * DD + t];
    s1[t] = fmaxf(a, 0.0f);
    __syncthreads();
    float b = bs2[t];
    for (int k = 0; k < DD; k++) b += s1[k] * Ws2[k * DD + t];
    summary2[t] = b;
}

// z = h * summary  -> bf16
__global__ void k_zmul(const float* h, const float* summary2, unsigned short* zbf,
                       long long total) {
    long long i = (long long)blockIdx.x * blockDim.x + threadIdx.x;
    if (i >= total) return;
    zbf[i] = f2bf(h[i] * summary2[i & (DD - 1)]);
}

// ---------------- batchnorm ----------------

__global__ void k_col_stats(const float* U, float* mean, float* inv, int N, int C) {
    __shared__ float sh[256];
    int c = blockIdx.x;
    float s = 0.0f, s2 = 0.0f;
    for (int n = threadIdx.x; n < N; n += blockDim.x) {
        float v = U[(long long)n * C + c];
        s += v; s2 += v * v;
    }
    sh[threadIdx.x] = s; __syncthreads();
    for (int t = 128; t > 0; t >>= 1) {
        if ((int)threadIdx.x < t) sh[threadIdx.x] += sh[threadIdx.x + t];
        __syncthreads();
    }
    float tot = sh[0]; __syncthreads();
    sh[threadIdx.x] = s2; __syncthreads();
    for (int t = 128; t > 0; t >>= 1) {
        if ((int)threadIdx.x < t) sh[threadIdx.x] += sh[threadIdx.x + t];
        __syncthreads();
    }
    if (threadIdx.x == 0) {
        float mu  = tot / N;
        float var = sh[0] / N - mu * mu;
        mean[c] = mu;
        inv[c]  = rsqrtf(var + EPSV);
    }
}

__global__ void k_bn_relu_bf16(const float* U, const float* mean, const float* inv,
                               const float* g, const float* be, unsigned short* outbf,
                               long long total, int C) {
    long long i = (long long)blockIdx.x * blockDim.x + threadIdx.x;
    if (i >= total) return;
    int c = (int)(i % C);
    float v = (U[i] - mean[c]) * inv[c] * g[c] + be[c];
    outbf[i] = f2bf(fmaxf(v, 0.0f));
}

// scores[n] = u[n,:] . Wd3 + bd3
__global__ void k_scores(const unsigned short* ubf_, const unsigned short* w3_,
                         const float* bd3, float* out, int N) {
    int n = blockIdx.x * blockDim.x + threadIdx.x;
    if (n >= N) return;
    const __bf16* u = (const __bf16*)ubf_ + (long long)n * DD;
    const __bf16* w = (const __bf16*)w3_;
    float s = bd3[0];
    for (int k = 0; k < DD; k++) s += (float)u[k] * (float)w[k];
    out[n] = s;
}

// ---------------- host launch ----------------

extern "C" void kernel_launch(void* const* d_in, const int* in_sizes, int n_in,
                              void* d_out, int out_size, void* d_ws, size_t ws_size,
                              hipStream_t stream) {
    const float* x    = (const float*)d_in[0];
    const int*   ei   = (const int*)  d_in[1];
    const float* Wg   = (const float*)d_in[2];
    const float* bg   = (const float*)d_in[3];
    const float* Ws1  = (const float*)d_in[4];
    const float* bs1  = (const float*)d_in[5];
    const float* Ws2  = (const float*)d_in[6];
    const float* bs2  = (const float*)d_in[7];
    const float* Wd1  = (const float*)d_in[8];
    const float* bd1  = (const float*)d_in[9];
    const float* g1   = (const float*)d_in[10];
    const float* be1  = (const float*)d_in[11];
    const float* Wd2  = (const float*)d_in[12];
    const float* bd2  = (const float*)d_in[13];
    const float* g2   = (const float*)d_in[14];
    const float* be2  = (const float*)d_in[15];
    const float* Wd3  = (const float*)d_in[16];
    const float* bd3  = (const float*)d_in[17];

    const long long N = in_sizes[0] / DD;     // 200000
    const long long E = in_sizes[1] / 2;      // 6400000

    // workspace carve-out
    char* ws = (char*)d_ws;
    size_t off = 0;
    float* bufA = (float*)(ws + off); off += (size_t)N * DD * 4;           // agg -> h -> u2(preBN)
    float* bufU = (float*)(ws + off); off += (size_t)N * HH2 * 4;          // u1(preBN)
    unsigned short* bufBf = (unsigned short*)(ws + off); off += (size_t)N * HH2 * 2; // bf16 activations
    float* deg    = (float*)(ws + off); off += (size_t)N * 4;
    float* logits = (float*)(ws + off); off += (size_t)N * 4;
    unsigned short* WgT  = (unsigned short*)(ws + off); off += DD * DD * 2;
    unsigned short* Wd1T = (unsigned short*)(ws + off); off += HH2 * DD * 2;
    unsigned short* Wd2T = (unsigned short*)(ws + off); off += DD * HH2 * 2;
    unsigned short* W3b  = (unsigned short*)(ws + off); off += DD * 2;
    float* partA    = (float*)(ws + off); off += NPART * 4;
    float* partB    = (float*)(ws + off); off += NPART * 4;
    float* scal     = (float*)(ws + off); off += 64;    // [0]=max, [1]=1/sumexp
    float* summary0 = (float*)(ws + off); off += DD * 4;
    float* summary2 = (float*)(ws + off); off += DD * 4;
    float* colmean  = (float*)(ws + off); off += HH2 * 4;
    float* colinv   = (float*)(ws + off); off += HH2 * 4;

    const int T = 256;
    auto cdiv = [](long long a, long long b) { return (unsigned)((a + b - 1) / b); };

    // 1. zero accumulators
    k_zero_f32<<<4096, T, 0, stream>>>(bufA, N * DD);
    k_zero_f32<<<64,   T, 0, stream>>>(deg, N);
    // 2. bf16-transpose weights
    k_conv_weights<<<1024, T, 0, stream>>>(Wg, Wd1, Wd2, Wd3, WgT, Wd1T, Wd2T, W3b);
    // 3. graph aggregation
    k_edge_deg<<<cdiv(E, T), T, 0, stream>>>(ei, deg, E);
    k_edge_scatter<<<cdiv(E * 64, T), T, 0, stream>>>(ei, x, bufA, E);
    // 4. m = x + agg/deg -> bf16
    k_prep_m<<<cdiv(N * DD, T), T, 0, stream>>>(x, bufA, deg, bufBf, N * DD);
    // 5. h = relu(m @ Wg + bg)   [N,256]  (overwrites agg buffer)
    {
        unsigned blocks = cdiv((N / 32) * (DD / 64), 8);
        k_wmma_gemm<<<blocks, T, 0, stream>>>(bufBf, WgT, bg, bufA, (int)N, DD, DD, 1);
    }
    // 6-8. attention pooling
    k_row_logits<<<cdiv(N, T), T, 0, stream>>>(bufA, logits, (int)N);
    k_partial_max<<<NPART, T, 0, stream>>>(logits, partA, (int)N);
    k_final_max<<<1, T, 0, stream>>>(partA, scal);
    k_partial_sumexp<<<NPART, T, 0, stream>>>(logits, scal, partB, (int)N);
    k_final_sumexp<<<1, T, 0, stream>>>(partB, scal);
    k_col_wsum<<<DD, T, 0, stream>>>(bufA, logits, scal, summary0, (int)N);
    // 9. summary MLP
    k_summary_mlp<<<1, T, 0, stream>>>(Ws1, bs1, Ws2, bs2, summary0, summary2);
    // 10. z = h * summary -> bf16
    k_zmul<<<cdiv(N * DD, T), T, 0, stream>>>(bufA, summary2, bufBf, N * DD);
    // 11. u1 = z @ Wd1 + bd1   [N,512]
    {
        unsigned blocks = cdiv((N / 32) * (HH2 / 64), 8);
        k_wmma_gemm<<<blocks, T, 0, stream>>>(bufBf, Wd1T, bd1, bufU, (int)N, DD, HH2, 0);
    }
    // 12-13. BN1 + relu -> bf16
    k_col_stats<<<HH2, T, 0, stream>>>(bufU, colmean, colinv, (int)N, HH2);
    k_bn_relu_bf16<<<cdiv(N * HH2, T), T, 0, stream>>>(bufU, colmean, colinv, g1, be1,
                                                       bufBf, N * HH2, HH2);
    // 14. u2 = u1a @ Wd2 + bd2  [N,256]  (reuse bufA)
    {
        unsigned blocks = cdiv((N / 32) * (DD / 64), 8);
        k_wmma_gemm<<<blocks, T, 0, stream>>>(bufBf, Wd2T, bd2, bufA, (int)N, HH2, DD, 0);
    }
    // 15-16. BN2 + relu -> bf16
    k_col_stats<<<DD, T, 0, stream>>>(bufA, colmean, colinv, (int)N, DD);
    k_bn_relu_bf16<<<cdiv(N * DD, T), T, 0, stream>>>(bufA, colmean, colinv, g2, be2,
                                                      bufBf, N * DD, DD);
    // 17. scores
    k_scores<<<cdiv(N, T), T, 0, stream>>>(bufBf, W3b, bd3, (float*)d_out, (int)N);
}
